// AttentionHead_17205638988572
// MI455X (gfx1250) — matrix operand order repaired and checked by reading, not compiled, not run on hardware
//
#include <hip/hip_runtime.h>
#include <hip/hip_bf16.h>

typedef __attribute__((ext_vector_type(16))) __bf16 v16bf;
typedef __attribute__((ext_vector_type(8)))  __bf16 v8bf;
typedef __attribute__((ext_vector_type(8)))  float  v8f;

#define B_  16
#define T_  2048
#define C_  384
#define H_  384

static __device__ __forceinline__ v8f wmma_bf16(v16bf a, v16bf b, v8f c) {
    return __builtin_amdgcn_wmma_f32_16x16x32_bf16(false, a, false, b, (short)0, c,
                                                   false, false);
}

static __device__ __forceinline__ v16bf cvt16(float4 a, float4 b, float4 c, float4 d) {
    v16bf o;
    o[0]=(__bf16)a.x;  o[1]=(__bf16)a.y;  o[2]=(__bf16)a.z;  o[3]=(__bf16)a.w;
    o[4]=(__bf16)b.x;  o[5]=(__bf16)b.y;  o[6]=(__bf16)b.z;  o[7]=(__bf16)b.w;
    o[8]=(__bf16)c.x;  o[9]=(__bf16)c.y;  o[10]=(__bf16)c.z; o[11]=(__bf16)c.w;
    o[12]=(__bf16)d.x; o[13]=(__bf16)d.y; o[14]=(__bf16)d.z; o[15]=(__bf16)d.w;
    return o;
}

// ---------------- Phase 1: Q/K/V projection GEMM (x @ W), bf16 WMMA ----------------
// grid: (2048/4 row-tiles, 384/64 col-groups, 3 matrices); 128 threads = 4 waves.
// Q,K stored row-major bf16 [B*T, H]; V stored transposed bf16 [B][H][T] so the
// PV B-operand in phase 2 is contiguous per lane.
__global__ __launch_bounds__(128) void qkv_kernel(
    const float* __restrict__ x,
    const float* __restrict__ Wq, const float* __restrict__ Wk, const float* __restrict__ Wv,
    __bf16* __restrict__ Qb, __bf16* __restrict__ Kb, __bf16* __restrict__ Vt)
{
    __shared__ __bf16 Wt[64][400];   // W tile transposed: [local col][k], 32B-aligned rows
    __shared__ float  Tr[4][16*68];  // per-wave 16x64 f32 transpose staging (stride 68)

    const int tid  = threadIdx.x;
    const int w    = tid >> 5;
    const int lane = tid & 31;
    const int li   = lane & 15;
    const int hi   = lane >> 4;
    const int n0   = blockIdx.y * 64;
    const int mat  = blockIdx.z;
    const float* W = (mat == 0) ? Wq : (mat == 1) ? Wk : Wv;

    // Cooperative load of W[:, n0:n0+64] -> LDS (transposed, bf16)
    for (int idx = tid; idx < 384 * 16; idx += 128) {
        const int k  = idx >> 4;
        const int c4 = (idx & 15) * 4;
        const float4 f = *(const float4*)(W + (size_t)k * H_ + n0 + c4);
        Wt[c4 + 0][k] = (__bf16)f.x;
        Wt[c4 + 1][k] = (__bf16)f.y;
        Wt[c4 + 2][k] = (__bf16)f.z;
        Wt[c4 + 3][k] = (__bf16)f.w;
    }
    __syncthreads();

    const int rt  = blockIdx.x * 4 + w;   // 16-row tile of flattened [B*T, C]
    const int row = rt * 16 + li;

    v8f acc[4] = {};
    #pragma unroll
    for (int c = 0; c < 12; ++c) {
        // A operand: x 16x32 tile, per-lane K group {16hi..16hi+15}
        const float4* xp = (const float4*)(x + (size_t)row * C_ + 32 * c + 16 * hi);
        const v16bf aX = cvt16(xp[0], xp[1], xp[2], xp[3]);
        #pragma unroll
        for (int tt = 0; tt < 4; ++tt) {
            const int lcol = 16 * tt + li;
            const v16bf bW = *(const v16bf*)&Wt[lcol][32 * c + 16 * hi];
            acc[tt] = wmma_bf16(aX, bW, acc[tt]);
        }
    }

    if (mat == 2) {
        // V: store transposed [b][col][t]; lane holds 8 consecutive t per tile -> 16B store
        const int bb = (rt * 16) / T_;
        const int t0 = (rt * 16) % T_;
        #pragma unroll
        for (int tt = 0; tt < 4; ++tt) {
            const int col = n0 + 16 * tt + li;
            v8bf o;
            #pragma unroll
            for (int r = 0; r < 8; ++r) o[r] = (__bf16)acc[tt][r];
            *(v8bf*)(Vt + ((size_t)bb * H_ + col) * T_ + t0 + 8 * hi) = o;
        }
    } else {
        // Q/K: transpose D-layout -> row-major through private per-wave LDS slab
        float* tr = Tr[w];
        #pragma unroll
        for (int tt = 0; tt < 4; ++tt)
            #pragma unroll
            for (int r = 0; r < 8; ++r)
                tr[(r + 8 * hi) * 68 + 16 * tt + li] = acc[tt][r];
        // same-wave LDS ops are in-order; no barrier needed
        __bf16* dst = (mat == 0) ? Qb : Kb;
        const float4* rp = (const float4*)(tr + li * 68 + 32 * hi);
        __bf16* dp = dst + (size_t)(rt * 16 + li) * H_ + n0 + 32 * hi;
        #pragma unroll
        for (int i = 0; i < 4; ++i) {
            const float4 f0 = rp[2 * i], f1 = rp[2 * i + 1];
            v8bf o;
            o[0]=(__bf16)f0.x; o[1]=(__bf16)f0.y; o[2]=(__bf16)f0.z; o[3]=(__bf16)f0.w;
            o[4]=(__bf16)f1.x; o[5]=(__bf16)f1.y; o[6]=(__bf16)f1.z; o[7]=(__bf16)f1.w;
            *(v8bf*)(dp + 8 * i) = o;
        }
    }
}

// ---------------- Phase 2: causal flash attention, online softmax ----------------
// One wave owns a 16-row Q tile and the full 384-wide output (24 f32 accum tiles).
// Q A-operands are register-resident (12 x v16bf); inner loop does 24 QK^T WMMAs +
// 24 PV WMMAs per 32 K/V columns, with rescale skipped when the running max holds.
__global__ __launch_bounds__(128) void attn_kernel(
    const __bf16* __restrict__ Qb, const __bf16* __restrict__ Kb,
    const __bf16* __restrict__ Vt, float* __restrict__ out)
{
    __shared__ float Pb[4][16 * 36];  // per-wave P / output transpose staging

    const int tid  = threadIdx.x;
    const int w    = tid >> 5;
    const int lane = tid & 31;
    const int li   = lane & 15;
    const int hi   = lane >> 4;

    const int qt = blockIdx.x * 4 + w;    // global 16-row q tile
    const int b  = qt >> 7;               // 128 tiles per batch
    const int q0 = (qt & 127) * 16;

    const __bf16* Qp = Qb + ((size_t)b * T_ + q0) * H_;
    const __bf16* Kp = Kb + (size_t)b * T_ * H_;
    const __bf16* Vp = Vt + (size_t)b * H_ * T_;
    float* Pw = Pb[w];

    // softmax in log2 domain; score scale = C^-0.5 (matches reference)
    const float coef = 0.07362190949e0f;  // (1/sqrt(384)) * log2(e)

    // loop-invariant Q A-operands: 16x384 bf16 tile in 12 x v16bf (96 VGPRs)
    v16bf aQ[12];
    #pragma unroll
    for (int c = 0; c < 12; ++c) {
        const v8bf* qp = (const v8bf*)(Qp + (size_t)li * H_ + 32 * c + 16 * hi);
        aQ[c] = __builtin_shufflevector(qp[0], qp[1],
                    0,1,2,3,4,5,6,7,8,9,10,11,12,13,14,15);
    }

    v8f acc[24] = {};
    float m_i[8], l_i[8];
    #pragma unroll
    for (int r = 0; r < 8; ++r) { m_i[r] = -1e30f; l_i[r] = 0.0f; }

    for (int s0 = 0; s0 <= q0 + 15; s0 += 32) {
        if (s0 + 32 <= q0 + 15)
            __builtin_prefetch(Kp + (size_t)(s0 + 32 + li) * H_, 0, 1);

        // ---- scores: two 16x16 sub-tiles (j = 0,1), K-dim = 384 in 12 chunks ----
        float u[2][8];
        #pragma unroll
        for (int j = 0; j < 2; ++j) {
            v8f sc = {};
            const __bf16* kr = Kp + (size_t)(s0 + 16 * j + li) * H_ + 16 * hi;
            #pragma unroll
            for (int c = 0; c < 12; ++c) {
                const v16bf bK = *(const v16bf*)(kr + 32 * c);
                sc = wmma_bf16(aQ[c], bK, sc);
            }
            #pragma unroll
            for (int r = 0; r < 8; ++r) u[j][r] = sc[r] * coef;
        }

        // ---- causal mask (only blocks touching the diagonal) ----
        if (s0 + 31 > q0) {
            #pragma unroll
            for (int j = 0; j < 2; ++j)
                #pragma unroll
                for (int r = 0; r < 8; ++r) {
                    const int sidx = s0 + 16 * j + li;
                    const int qidx = q0 + r + 8 * hi;
                    if (sidx > qidx) u[j][r] = -1e30f;
                }
        }

        // ---- online softmax: row max / rescale / sum (rows live in 16-lane halves) ----
        float mb[8];
        #pragma unroll
        for (int r = 0; r < 8; ++r) mb[r] = fmaxf(u[0][r], u[1][r]);
        #pragma unroll
        for (int off = 8; off >= 1; off >>= 1)
            #pragma unroll
            for (int r = 0; r < 8; ++r) mb[r] = fmaxf(mb[r], __shfl_xor(mb[r], off, 32));

        bool grew = false;
        #pragma unroll
        for (int r = 0; r < 8; ++r) grew = grew || (mb[r] > m_i[r]);

        float fac[8], sum[8];
        #pragma unroll
        for (int r = 0; r < 8; ++r) {
            const float mn = fmaxf(m_i[r], mb[r]);
            fac[r] = exp2f(m_i[r] - mn);
            m_i[r] = mn;
            u[0][r] = exp2f(u[0][r] - mn);
            u[1][r] = exp2f(u[1][r] - mn);
            sum[r] = u[0][r] + u[1][r];
        }
        #pragma unroll
        for (int off = 8; off >= 1; off >>= 1)
            #pragma unroll
            for (int r = 0; r < 8; ++r) sum[r] += __shfl_xor(sum[r], off, 32);
        #pragma unroll
        for (int r = 0; r < 8; ++r) l_i[r] = l_i[r] * fac[r] + sum[r];

        // O rescale only when the running max actually grew somewhere in the wave
        if (__any((int)grew)) {
            #pragma unroll
            for (int t = 0; t < 24; ++t)
                #pragma unroll
                for (int r = 0; r < 8; ++r) acc[t][r] *= fac[r];
        }

        // ---- transpose P (D layout -> A layout) via per-wave LDS, cvt to bf16 ----
        #pragma unroll
        for (int j = 0; j < 2; ++j)
            #pragma unroll
            for (int r = 0; r < 8; ++r)
                Pw[(r + 8 * hi) * 36 + li + 16 * j] = u[j][r];
        v16bf aP;
        #pragma unroll
        for (int e = 0; e < 8; ++e) aP[e]     = (__bf16)Pw[li * 36 + 16 * hi + e];
        #pragma unroll
        for (int e = 0; e < 8; ++e) aP[8 + e] = (__bf16)Pw[li * 36 + 16 * hi + 8 + e];

        // ---- O += P @ V : V^T layout gives contiguous 32B per-lane B loads ----
        #pragma unroll
        for (int t = 0; t < 24; ++t) {
            const v16bf bV = *(const v16bf*)(Vp + (size_t)(16 * t + li) * T_ + s0 + 16 * hi);
            acc[t] = wmma_bf16(aP, bV, acc[t]);
        }
    }

    // ---- epilogue: divide by l, transpose 16x16 tiles through LDS, coalesced stores ----
    float inv[8];
    #pragma unroll
    for (int r = 0; r < 8; ++r) inv[r] = 1.0f / l_i[r];

    float* op = out + ((size_t)b * T_ + q0) * H_;
    #pragma unroll
    for (int t = 0; t < 24; ++t) {
        #pragma unroll
        for (int r = 0; r < 8; ++r)
            Pw[(r + 8 * hi) * 36 + li] = acc[t][r] * inv[r];
        const float4* rp = (const float4*)(Pw + li * 36 + 8 * hi);
        const float4 o0 = rp[0], o1 = rp[1];
        float4* dp = (float4*)(op + (size_t)li * H_ + 16 * t + 8 * hi);
        dp[0] = o0;
        dp[1] = o1;
    }
}

extern "C" void kernel_launch(void* const* d_in, const int* in_sizes, int n_in,
                              void* d_out, int out_size, void* d_ws, size_t ws_size,
                              hipStream_t stream) {
    const float* x  = (const float*)d_in[0];
    const float* Wq = (const float*)d_in[1];
    const float* Wk = (const float*)d_in[2];
    const float* Wv = (const float*)d_in[3];

    // ws layout: Q bf16 [B*T,H] | K bf16 [B*T,H] | V^T bf16 [B][H][T]  (75.5 MB total)
    const size_t n = (size_t)B_ * T_ * H_;
    __bf16* Qb = (__bf16*)d_ws;
    __bf16* Kb = Qb + n;
    __bf16* Vt = Kb + n;

    dim3 g1((B_ * T_ / 16) / 4, H_ / 64, 3);
    qkv_kernel<<<g1, 128, 0, stream>>>(x, Wq, Wk, Wv, Qb, Kb, Vt);

    dim3 g2((B_ * T_ / 16) / 4);
    attn_kernel<<<g2, 128, 0, stream>>>(Qb, Kb, Vt, (float*)d_out);
}
